// OOKB_30623116821277
// MI455X (gfx1250) — matrix operand order repaired and checked by reading, not compiled, not run on hardware
//
#include <hip/hip_runtime.h>
#include <stdint.h>

#define N_ENT  50000
#define N_REL2 16
#define EMB    128
#define N_EDGE 800000
#define K_DIM  2048   // N_REL2 * EMB

typedef __attribute__((ext_vector_type(16))) __bf16 v16bf;
typedef __attribute__((ext_vector_type(8)))  float  v8f;

union AF { uint4 q[2]; v16bf v; };

// float -> bf16 round-to-nearest-even (bit trick, no header dependency)
static __device__ __forceinline__ unsigned f2bf_rne(float f) {
  unsigned u = __float_as_uint(f);
  unsigned r = (u + 0x7FFFu + ((u >> 16) & 1u)) >> 16;
  return r & 0xFFFFu;
}
static __device__ __forceinline__ unsigned pack2bf(float a, float b) {
  return f2bf_rne(a) | (f2bf_rne(b) << 16);
}

// ---------------------------------------------------------------------------
// Kernel 1: zero agg (bf16) + deg, and build Wt in bf16 with layout
// wtb[f * 2048 + (r*128 + d)] = rel_weight[r][f][d]  (k contiguous per f row)
// ---------------------------------------------------------------------------
__global__ void OOKB_init_kernel(uint4* __restrict__ agg4, float* __restrict__ deg,
                                 unsigned short* __restrict__ wtb,
                                 const float* __restrict__ rw) {
  const uint4 z = make_uint4(0u, 0u, 0u, 0u);
  size_t stride = (size_t)gridDim.x * blockDim.x;
  size_t tid = (size_t)blockIdx.x * blockDim.x + threadIdx.x;
  const size_t AGG_Q = (size_t)N_ENT * N_REL2 * EMB / 8;   // 12,800,000 uint4
  for (size_t i = tid; i < AGG_Q; i += stride) agg4[i] = z;
  for (size_t i = tid; i < (size_t)N_ENT; i += stride) deg[i] = 0.0f;
  for (size_t i = tid; i < (size_t)EMB * K_DIM; i += stride) {
    unsigned f = (unsigned)(i >> 11);          // 0..127
    unsigned k = (unsigned)(i & 2047u);        // r*128 + d
    unsigned r = k >> 7, d = k & 127u;
    wtb[i] = (unsigned short)f2bf_rne(rw[((size_t)r * EMB + f) * EMB + d]);
  }
}

// ---------------------------------------------------------------------------
// Kernel 2: edge scatter. One wave32 per edge; lane handles 4 features.
// agg[col*2048 + t*128 + d] += bf16(x0[row][d])  via global_atomic_pk_add_bf16
// ---------------------------------------------------------------------------
__global__ void __launch_bounds__(256)
OOKB_scatter_kernel(const float* __restrict__ emb, const float* __restrict__ km,
                    const int* __restrict__ erow, const int* __restrict__ ecol,
                    const int* __restrict__ etype,
                    unsigned long long aggBase, float* __restrict__ deg) {
  int e = blockIdx.x * 8 + (threadIdx.x >> 5);
  if (e >= N_EDGE) return;
  int lane = threadIdx.x & 31;
  int row = erow[e];
  int col = ecol[e];
  int t   = etype[e];
  if (lane == 0) atomicAdd(deg + col, 1.0f);       // deg counts every edge
  float mk = km[row];
  if (mk == 0.0f) return;                          // masked-out source: zero payload
  const float4* er = (const float4*)(emb + (size_t)row * EMB);
  float4 v = er[lane];
  v.x *= mk; v.y *= mk; v.z *= mk; v.w *= mk;
  unsigned p0 = pack2bf(v.x, v.y);
  unsigned p1 = pack2bf(v.z, v.w);
  unsigned boff = ((unsigned)col * (unsigned)K_DIM +
                   (unsigned)t * (unsigned)EMB + (unsigned)lane * 4u) * 2u;
  asm volatile("global_atomic_pk_add_bf16 %0, %1, %2"
               :: "v"(boff), "v"(p0), "s"(aggBase) : "memory");
  asm volatile("global_atomic_pk_add_bf16 %0, %1, %2 offset:4"
               :: "v"(boff), "v"(p1), "s"(aggBase) : "memory");
}

// ---------------------------------------------------------------------------
// Kernel 3: out = Agg(N x 2048, bf16) @ Wt(2048 x 128, bf16) with
// v_wmma_f32_16x16x32_bf16, fused deg/mask/L2-normalize epilogue.
// Block = 16 rows x 128 cols; 4 waves, each owns TWO 16x16 column tiles
// (cols [32w, 32w+32)) so the A fragment is loaded once and fed to 2 WMMAs.
// Fragment layouts (contiguous k order used here, same for A and B):
//   A lane<16 : M=lane,    K = k0+0..15
//   A lane>=16: M=lane-16, K = k0+16..31
//   B lane<16 : N=lane,    K = k0+0..15 ; lane>=16: N=lane-16, K = k0+16..31
// Every fragment = two global_load_b128 per lane.
// ---------------------------------------------------------------------------
__global__ void __launch_bounds__(128)
OOKB_gemm_kernel(const uint4* __restrict__ agg4, const uint4* __restrict__ wt4,
                 const float* __restrict__ deg, const float* __restrict__ km,
                 const float* __restrict__ emb, float* __restrict__ out) {
  __shared__ float rowsq[16];
  __shared__ float lrdg[16];   // rcp(max(deg,1)) per block row
  __shared__ float lmk[16];    // known_mask per block row
  int tid = threadIdx.x;
  int R0 = blockIdx.x * 16;
  if (tid < 16) {
    rowsq[tid] = 0.0f;
    lrdg[tid]  = __builtin_amdgcn_rcpf(fmaxf(deg[R0 + tid], 1.0f));
    lmk[tid]   = km[R0 + tid];
  }
  __syncthreads();

  int lane = tid & 31, wave = tid >> 5;     // 4 waves
  int col  = lane & 15, hf = lane >> 4;
  int F0 = wave * 32;                       // two 16-col tiles per wave

  // uint4 = 8 bf16 elements
  const uint4* aP  = agg4 + (size_t)(R0 + col) * (K_DIM / 8) + hf * 2;       // A tile
  const uint4* bP0 = wt4  + (size_t)(F0 + col) * (K_DIM / 8) + hf * 2;       // B tile 0
  const uint4* bP1 = wt4  + (size_t)(F0 + 16 + col) * (K_DIM / 8) + hf * 2;  // B tile 1

  v8f c0 = {0.f, 0.f, 0.f, 0.f, 0.f, 0.f, 0.f, 0.f};
  v8f c1 = {0.f, 0.f, 0.f, 0.f, 0.f, 0.f, 0.f, 0.f};
#pragma unroll 2
  for (int k0 = 0; k0 < K_DIM; k0 += 32) {
    AF a, b0, b1;
    a.q[0]  = aP[0];   // K = kA .. kA+7
    a.q[1]  = aP[1];   // K = kA+8 .. kA+15
    b0.q[0] = bP0[0];  // K = kB .. kB+7
    b0.q[1] = bP0[1];  // K = kB+8 .. kB+15
    b1.q[0] = bP1[0];
    b1.q[1] = bP1[1];
    aP += 4; bP0 += 4; bP1 += 4;  // advance 32 bf16
    c0 = __builtin_amdgcn_wmma_f32_16x16x32_bf16(false, a.v, false, b0.v,
                                                 (short)0, c0, false, false);
    c1 = __builtin_amdgcn_wmma_f32_16x16x32_bf16(false, a.v, false, b1.v,
                                                 (short)0, c1, false, false);
  }

  // Epilogue: VGPR j holds row m = j + 8*hf, columns F0+col and F0+16+col
  float h0[8], h1[8];
#pragma unroll
  for (int j = 0; j < 8; ++j) {
    int m = j + hf * 8;
    int n = R0 + m;
    float rdg = lrdg[m];
    float mk  = lmk[m];
    float om  = 1.0f - mk;
    float em0 = emb[(size_t)n * EMB + F0 + col];
    float em1 = emb[(size_t)n * EMB + F0 + 16 + col];
    h0[j] = mk * em0 + om * (c0[j] * rdg);   // mask is 0/1: mk*x0 == mk*emb
    h1[j] = mk * em1 + om * (c1[j] * rdg);
    float s = h0[j] * h0[j] + h1[j] * h1[j];
    s += __shfl_xor(s, 1);   // masks <16 stay within each 16-lane half-group
    s += __shfl_xor(s, 2);
    s += __shfl_xor(s, 4);
    s += __shfl_xor(s, 8);
    if (col == j) atomicAdd(&rowsq[m], s);   // one lane per half-group per row
  }
  __syncthreads();
#pragma unroll
  for (int j = 0; j < 8; ++j) {
    int m = j + hf * 8;
    int n = R0 + m;
    // 1/max(sqrt(s),1e-12): rsq + clamp (s==0 -> inf -> clamp to 1e12)
    float inv = fminf(__builtin_amdgcn_rsqf(rowsq[m]), 1e12f);
    out[(size_t)n * EMB + F0 + col]      = h0[j] * inv;
    out[(size_t)n * EMB + F0 + 16 + col] = h1[j] * inv;
  }
}

// ---------------------------------------------------------------------------
extern "C" void kernel_launch(void* const* d_in, const int* in_sizes, int n_in,
                              void* d_out, int out_size, void* d_ws, size_t ws_size,
                              hipStream_t stream) {
  const float* emb = (const float*)d_in[0];   // (50000,128) f32
  const float* rw  = (const float*)d_in[1];   // (16,128,128) f32
  const float* km  = (const float*)d_in[2];   // (50000,) f32
  const int*   ei  = (const int*)d_in[3];     // (2,800000) i32
  const int*   et  = (const int*)d_in[4];     // (800000,) i32
  float* out = (float*)d_out;                 // (50000,128) f32

  char* ws = (char*)d_ws;
  const size_t AGG_BYTES = (size_t)N_ENT * N_REL2 * EMB * 2;  // 204,800,000 (bf16)
  uint4* agg4 = (uint4*)ws;
  float* deg  = (float*)(ws + AGG_BYTES);
  unsigned short* wtb = (unsigned short*)(ws + AGG_BYTES + (size_t)N_ENT * sizeof(float));

  OOKB_init_kernel<<<2048, 256, 0, stream>>>(agg4, deg, wtb, rw);
  OOKB_scatter_kernel<<<N_EDGE / 8, 256, 0, stream>>>(
      emb, km, ei, ei + N_EDGE, et, (unsigned long long)(uintptr_t)ws, deg);
  OOKB_gemm_kernel<<<N_ENT / 16, 128, 0, stream>>>(
      agg4, (const uint4*)wtb, deg, km, emb, out);
}
